// tsfefbs_13056700580031
// MI455X (gfx1250) — matrix operand order repaired and checked by reading, not compiled, not run on hardware
//
#include <hip/hip_runtime.h>
#include <hip/hip_bf16.h>
#include <math.h>

typedef __attribute__((ext_vector_type(16))) __bf16       v16bf;
typedef __attribute__((ext_vector_type(8)))  __bf16       v8bf;
typedef __attribute__((ext_vector_type(8)))  float        v8f;
typedef __attribute__((ext_vector_type(8)))  unsigned int v8ui;

#define WPB 8     // waves per block (wave32)
#define TPB 256

#define BB 4
#define CC 64
#define HH 160
#define WW 160
#define HW (HH*WW)                 // 25600
#define SZ ((size_t)BB*CC*HW)      // 6,553,600 f32 elements
#define VP 1156                    // 34*34 padded view
#define YP 170                     // 160 + 2*5 halo

__device__ __forceinline__ float lrelu(float v){ return v >= 0.f ? v : 0.1f*v; }

// 16-bit A-matrix 16x32 element map (contiguous: K = 16*hi + e). lane half hi (0/1), elem e -> K.
__device__ __forceinline__ int a_kmap(int e, int hi){
  return (hi << 4) + e;
}

__device__ __forceinline__ v16bf cvt16(float4 a, float4 b, float4 c, float4 d){
  v16bf r;
  r[0]=(__bf16)a.x;  r[1]=(__bf16)a.y;  r[2]=(__bf16)a.z;  r[3]=(__bf16)a.w;
  r[4]=(__bf16)b.x;  r[5]=(__bf16)b.y;  r[6]=(__bf16)b.z;  r[7]=(__bf16)b.w;
  r[8]=(__bf16)c.x;  r[9]=(__bf16)c.y;  r[10]=(__bf16)c.z; r[11]=(__bf16)c.w;
  r[12]=(__bf16)d.x; r[13]=(__bf16)d.y; r[14]=(__bf16)d.z; r[15]=(__bf16)d.w;
  return r;
}

// packed 32-bit zero-select (8x v_cndmask_b32, dual-issuable)
__device__ __forceinline__ v16bf zsel(v16bf x, bool valid){
  v8ui t = __builtin_bit_cast(v8ui, x);
  v8ui z = {};
  t = valid ? t : z;
  return __builtin_bit_cast(v16bf, t);
}

// ---------------------------------------------------------------------------
// NCHW f32 -> NHWC f32, LDS-tiled 64x64 transpose (pad 65: no bank conflicts)
// ---------------------------------------------------------------------------
__global__ void k_tr_in(const float* __restrict__ in, float* __restrict__ out)
{
  __shared__ float t[64 * 65];
  const int bb   = blockIdx.x / 400;
  const int tile = blockIdx.x % 400;
  const size_t hw0 = (size_t)tile * 64;
  const int cg  = threadIdx.x >> 6;   // 0..3
  const int lo  = threadIdx.x & 63;
  #pragma unroll 4
  for (int i = 0; i < 16; ++i) {
    int c = i * 4 + cg;
    t[c * 65 + lo] = in[((size_t)bb * CC + c) * HW + hw0 + lo];
  }
  __syncthreads();
  #pragma unroll 4
  for (int i = 0; i < 16; ++i) {
    int hwl = i * 4 + cg;
    out[((size_t)bb * HW + hw0 + hwl) * CC + lo] = t[lo * 65 + hwl];
  }
}

// NCHW f32 -> per-view padded SAI layout (25 views of 34x34, 1-px zero halo)
__global__ void k_tr_in_sai(const float* __restrict__ in, float* __restrict__ outp)
{
  __shared__ float t[64 * 65];
  const int bb   = blockIdx.x / 400;
  const int tile = blockIdx.x % 400;
  const size_t hw0 = (size_t)tile * 64;
  const int cg  = threadIdx.x >> 6;
  const int lo  = threadIdx.x & 63;
  #pragma unroll 4
  for (int i = 0; i < 16; ++i) {
    int c = i * 4 + cg;
    t[c * 65 + lo] = in[((size_t)bb * CC + c) * HW + hw0 + lo];
  }
  __syncthreads();
  #pragma unroll 4
  for (int i = 0; i < 16; ++i) {
    int hw = (int)hw0 + i * 4 + cg;
    int h = hw / WW, w = hw % WW;
    int bi = h >> 5, ii = h & 31, bj = w >> 5, jj = w & 31;
    outp[(((size_t)bb * 25 + bi * 5 + bj) * VP + (ii + 1) * 34 + (jj + 1)) * CC + lo]
        = t[lo * 65 + (i * 4 + cg)];
  }
}

// NHWC f32 -> NCHW f32
__global__ void k_tr_out(const float* __restrict__ in, float* __restrict__ out)
{
  __shared__ float t[64 * 65];
  const int bb   = blockIdx.x / 400;
  const int tile = blockIdx.x % 400;
  const size_t hw0 = (size_t)tile * 64;
  const int cg  = threadIdx.x >> 6;
  const int lo  = threadIdx.x & 63;
  #pragma unroll 4
  for (int i = 0; i < 16; ++i) {
    int hwl = i * 4 + cg;
    t[lo * 65 + hwl] = in[((size_t)bb * HW + hw0 + hwl) * CC + lo];
  }
  __syncthreads();
  #pragma unroll 4
  for (int i = 0; i < 16; ++i) {
    int c = i * 4 + cg;
    out[((size_t)bb * CC + c) * HW + hw0 + lo] = t[c * 65 + lo];
  }
}

// ---------------------------------------------------------------------------
// Repack f32 OIHW conv weights into bf16, per-lane WMMA A-matrix element order:
// flat q = (((coT*taps + tap)*2 + kk2)*32 + lane)*16 + e
// ---------------------------------------------------------------------------
__global__ void k_pack(const float* __restrict__ src, __bf16* __restrict__ dst,
                       int taps, int total)
{
  int q = blockIdx.x * blockDim.x + threadIdx.x;
  if (q >= total) return;
  int e    = q & 15;
  int lane = (q >> 4) & 31;
  int kk2  = (q >> 9) & 1;
  int rest = q >> 10;
  int tap  = rest % taps, coT = rest / taps;
  int m = lane & 15, hi = lane >> 4;
  int ci = kk2 * 32 + a_kmap(e, hi);
  int co = coT * 16 + m;
  dst[q] = (__bf16)src[((size_t)co * CC + ci) * taps + tap];
}

// ---------------------------------------------------------------------------
// block_s: 3x3 conv pad 1 on 32x32 views, input in padded-view layout (no
// selects; taps are immediate offsets from a per-lane base). lrelu -> cat br 0.
// ---------------------------------------------------------------------------
__global__ void k_conv_spa(const float* __restrict__ srcp,   // padded views f32
                           const __bf16* __restrict__ wpk,
                           __bf16* __restrict__ cat)
{
  const int lane = threadIdx.x & 31;
  const int wave = blockIdx.x * WPB + (threadIdx.x >> 5);
  const int tilesPerB = 4 * 800;
  const int b    = wave / tilesPerB;
  const int r0   = wave % tilesPerB;
  const int coT  = r0 / 800;
  const int pixT = r0 % 800;
  const int nIdx = lane & 15;
  const int hi   = lane >> 4;

  int hdst[2], wdst[2];
  const float* base[2];
  #pragma unroll
  for (int u = 0; u < 2; ++u) {
    int p = pixT * 32 + nIdx + u * 16;
    int view = p >> 10; int pp = p & 1023;
    int bi = view / 5, bj = view % 5;
    int ii = pp >> 5, jj = pp & 31;
    hdst[u] = ii * 5 + bi; wdst[u] = jj * 5 + bj;
    base[u] = srcp + (((size_t)b * 25 + view) * VP + (ii + 1) * 34 + (jj + 1)) * CC;
  }
  const __bf16* wl = wpk + ((size_t)coT * 9 * 2 * 32 + lane) * 16;

  v8f acc0 = {}, acc1 = {};
  #pragma unroll
  for (int tap = 0; tap < 9; ++tap) {
    const int off = ((tap / 3 - 1) * 34 + (tap % 3 - 1)) * CC;
    #pragma unroll
    for (int kk2 = 0; kk2 < 2; ++kk2) {
      v16bf Am = *(const v16bf*)(wl + (tap * 2 + kk2) * 512);
      int ci0 = kk2 * 32 + (hi << 4);
      const float4* v0 = (const float4*)(base[0] + off + ci0);
      const float4* v1 = (const float4*)(base[1] + off + ci0);
      v16bf B0 = cvt16(v0[0], v0[1], v0[2], v0[3]);
      v16bf B1 = cvt16(v1[0], v1[1], v1[2], v1[3]);
      acc0 = __builtin_amdgcn_wmma_f32_16x16x32_bf16(false, Am, false, B0,
                                                     (short)0, acc0, false, false);
      acc1 = __builtin_amdgcn_wmma_f32_16x16x32_bf16(false, Am, false, B1,
                                                     (short)0, acc1, false, false);
    }
  }
  const int c0 = coT * 16 + 8 * hi;
  v8bf o0, o1;
  #pragma unroll
  for (int r = 0; r < 8; ++r) { o0[r] = (__bf16)lrelu(acc0[r]); o1[r] = (__bf16)lrelu(acc1[r]); }
  *(v8bf*)(cat + (((size_t)b * HH + hdst[0]) * WW + wdst[0]) * 256 + c0) = o0;
  *(v8bf*)(cat + (((size_t)b * HH + hdst[1]) * WW + wdst[1]) * 256 + c0) = o1;
}

// ---------------------------------------------------------------------------
// block_a: 3x3 conv pad 1 on 5x5 macropixels, NHWC input, packed selects.
// ---------------------------------------------------------------------------
__global__ void k_conv_ang(const float* __restrict__ src,    // NHWC f32
                           const __bf16* __restrict__ wpk,
                           __bf16* __restrict__ cat)
{
  const int lane = threadIdx.x & 31;
  const int wave = blockIdx.x * WPB + (threadIdx.x >> 5);
  const int tilesPerB = 4 * 800;
  const int b    = wave / tilesPerB;
  const int r0   = wave % tilesPerB;
  const int coT  = r0 / 800;
  const int pixT = r0 % 800;
  const int nIdx = lane & 15;
  const int hi   = lane >> 4;

  int hdst[2], wdst[2], hb[2], vb[2], li[2], lj[2];
  #pragma unroll
  for (int u = 0; u < 2; ++u) {
    int p = pixT * 32 + nIdx + u * 16;
    int patch = p / 25; int r = p % 25;
    int ph = patch >> 5, pw = patch & 31;
    li[u] = r / 5; lj[u] = r % 5;
    hb[u] = ph * 5; vb[u] = pw * 5;
    hdst[u] = hb[u] + li[u]; wdst[u] = vb[u] + lj[u];
  }
  const float* srcb = src + (size_t)b * HW * CC;
  const __bf16* wl  = wpk + ((size_t)coT * 9 * 2 * 32 + lane) * 16;

  v8f acc0 = {}, acc1 = {};
  for (int tap = 0; tap < 9; ++tap) {
    int dy = tap / 3 - 1, dx = tap % 3 - 1;
    bool valid[2]; const float* bp[2];
    #pragma unroll
    for (int u = 0; u < 2; ++u) {
      int i2 = li[u] + dy, j2 = lj[u] + dx;
      valid[u] = (i2 >= 0 && i2 < 5 && j2 >= 0 && j2 < 5);
      int ic = i2 < 0 ? 0 : (i2 >= 5 ? 4 : i2);
      int jc = j2 < 0 ? 0 : (j2 >= 5 ? 4 : j2);
      bp[u] = srcb + ((size_t)(hb[u] + ic) * WW + (vb[u] + jc)) * CC;
    }
    #pragma unroll
    for (int kk2 = 0; kk2 < 2; ++kk2) {
      v16bf Am = *(const v16bf*)(wl + (tap * 2 + kk2) * 512);
      int ci0 = kk2 * 32 + (hi << 4);
      const float4* v0 = (const float4*)(bp[0] + ci0);
      const float4* v1 = (const float4*)(bp[1] + ci0);
      v16bf B0 = zsel(cvt16(v0[0], v0[1], v0[2], v0[3]), valid[0]);
      v16bf B1 = zsel(cvt16(v1[0], v1[1], v1[2], v1[3]), valid[1]);
      acc0 = __builtin_amdgcn_wmma_f32_16x16x32_bf16(false, Am, false, B0,
                                                     (short)0, acc0, false, false);
      acc1 = __builtin_amdgcn_wmma_f32_16x16x32_bf16(false, Am, false, B1,
                                                     (short)0, acc1, false, false);
    }
  }
  const int c0 = coT * 16 + 8 * hi;
  v8bf o0, o1;
  #pragma unroll
  for (int r = 0; r < 8; ++r) { o0[r] = (__bf16)lrelu(acc0[r]); o1[r] = (__bf16)lrelu(acc1[r]); }
  *(v8bf*)(cat + (((size_t)b * HH + hdst[0]) * WW + wdst[0]) * 256 + 64 + c0) = o0;
  *(v8bf*)(cat + (((size_t)b * HH + hdst[1]) * WW + wdst[1]) * 256 + 64 + c0) = o1;
}

// ---------------------------------------------------------------------------
// EPI stage 1: [1,15] kernel, stride 5, pad 5 along W (or H if vert). NHWC.
// ---------------------------------------------------------------------------
__global__ void k_epi1(const float* __restrict__ src,    // NHWC f32
                       const __bf16* __restrict__ wpk,   // packed (4coT,15taps)
                       __bf16* __restrict__ tmp,         // (B,160,32,64) bf16
                       int vert)
{
  const int lane = threadIdx.x & 31;
  const int wave = blockIdx.x * WPB + (threadIdx.x >> 5);
  const int tilesPerB = 4 * 160;
  const int b    = wave / tilesPerB;
  const int r0   = wave % tilesPerB;
  const int coT  = r0 / 160;
  const int pixT = r0 % 160;
  const int nIdx = lane & 15, hi = lane >> 4;

  int row[2], wa[2];
  #pragma unroll
  for (int u = 0; u < 2; ++u) {
    int p = pixT * 32 + nIdx + u * 16;
    row[u] = p >> 5; wa[u] = p & 31;
  }
  const float* srcb = src + (size_t)b * HW * CC;
  const __bf16* wl  = wpk + ((size_t)coT * 15 * 2 * 32 + lane) * 16;

  v8f acc0 = {}, acc1 = {};
  for (int t = 0; t < 15; ++t) {
    bool valid[2]; const float* bp[2];
    #pragma unroll
    for (int u = 0; u < 2; ++u) {
      int wsc = wa[u] * 5 - 5 + t;
      valid[u] = (wsc >= 0 && wsc < WW);
      int wc = wsc < 0 ? 0 : (wsc >= WW ? WW - 1 : wsc);
      bp[u] = srcb + (vert ? ((size_t)wc * WW + row[u])
                           : ((size_t)row[u] * WW + wc)) * CC;
    }
    #pragma unroll
    for (int kk2 = 0; kk2 < 2; ++kk2) {
      v16bf Am = *(const v16bf*)(wl + (t * 2 + kk2) * 512);
      int ci0 = kk2 * 32 + (hi << 4);
      const float4* v0 = (const float4*)(bp[0] + ci0);
      const float4* v1 = (const float4*)(bp[1] + ci0);
      v16bf B0 = zsel(cvt16(v0[0], v0[1], v0[2], v0[3]), valid[0]);
      v16bf B1 = zsel(cvt16(v1[0], v1[1], v1[2], v1[3]), valid[1]);
      acc0 = __builtin_amdgcn_wmma_f32_16x16x32_bf16(false, Am, false, B0,
                                                     (short)0, acc0, false, false);
      acc1 = __builtin_amdgcn_wmma_f32_16x16x32_bf16(false, Am, false, B1,
                                                     (short)0, acc1, false, false);
    }
  }
  const int c0 = coT * 16 + 8 * hi;
  v8bf o0, o1;
  #pragma unroll
  for (int r = 0; r < 8; ++r) { o0[r] = (__bf16)lrelu(acc0[r]); o1[r] = (__bf16)lrelu(acc1[r]); }
  *(v8bf*)(tmp + (((size_t)b * HH + row[0]) * 32 + wa[0]) * CC + c0) = o0;
  *(v8bf*)(tmp + (((size_t)b * HH + row[1]) * 32 + wa[1]) * CC + c0) = o1;
}

// ---------------------------------------------------------------------------
// EPI stage 2: 1x1 conv 64 -> 320, lrelu, pixel shuffle into concat branch.
// ---------------------------------------------------------------------------
__global__ void k_epi2(const __bf16* __restrict__ tmp,   // (B,160,32,64) bf16
                       const __bf16* __restrict__ wpk,   // packed (20coT,1tap)
                       __bf16* __restrict__ cat,
                       int vert)
{
  const int lane = threadIdx.x & 31;
  const int wave = blockIdx.x * WPB + (threadIdx.x >> 5);
  const int tilesPerB = 20 * 160;
  const int b    = wave / tilesPerB;
  const int r0   = wave % tilesPerB;
  const int coT  = r0 / 160;
  const int pixT = r0 % 160;
  const int nIdx = lane & 15, hi = lane >> 4;

  int row[2], wa[2];
  const __bf16* bp[2];
  #pragma unroll
  for (int u = 0; u < 2; ++u) {
    int p = pixT * 32 + nIdx + u * 16;
    row[u] = p >> 5; wa[u] = p & 31;
    bp[u] = tmp + (((size_t)b * HH + row[u]) * 32 + wa[u]) * CC;
  }
  const __bf16* wl = wpk + ((size_t)coT * 2 * 32 + lane) * 16;

  v8f acc0 = {}, acc1 = {};
  #pragma unroll
  for (int kk2 = 0; kk2 < 2; ++kk2) {
    v16bf Am = *(const v16bf*)(wl + kk2 * 512);
    int ci0 = kk2 * 32 + (hi << 4);
    v16bf B0 = *(const v16bf*)(bp[0] + ci0);
    v16bf B1 = *(const v16bf*)(bp[1] + ci0);
    acc0 = __builtin_amdgcn_wmma_f32_16x16x32_bf16(false, Am, false, B0,
                                                   (short)0, acc0, false, false);
    acc1 = __builtin_amdgcn_wmma_f32_16x16x32_bf16(false, Am, false, B1,
                                                   (short)0, acc1, false, false);
  }
  const int oc0 = coT * 16 + 8 * hi;       // 8 consecutive out-channels
  const int t   = oc0 >> 6, c0 = oc0 & 63; // same shuffle phase t for all 8
  v8bf o0, o1;
  #pragma unroll
  for (int r = 0; r < 8; ++r) { o0[r] = (__bf16)lrelu(acc0[r]); o1[r] = (__bf16)lrelu(acc1[r]); }
  const int bro = vert ? 192 : 128;
  #pragma unroll
  for (int u = 0; u < 2; ++u) {
    size_t pix = vert ? (((size_t)b * HH + (wa[u] * 5 + t)) * WW + row[u])
                      : (((size_t)b * HH + row[u]) * WW + (wa[u] * 5 + t));
    *(v8bf*)(cat + pix * 256 + bro + c0) = (u == 0) ? o0 : o1;
  }
}

// ---------------------------------------------------------------------------
// Attention statistics: per batch, 4 branch sums + 10 Gram entries.
// ---------------------------------------------------------------------------
__global__ void k_zero(float* p, int n){
  int i = blockIdx.x * blockDim.x + threadIdx.x;
  if (i < n) p[i] = 0.f;
}

__global__ void k_stats(const __bf16* __restrict__ cat, float* __restrict__ stats)
{
  const int b   = blockIdx.x >> 7;
  const int tid = (blockIdx.x & 127) * blockDim.x + threadIdx.x;
  const int M   = CC * HW;
  float s[4]  = {0.f, 0.f, 0.f, 0.f};
  float g[10] = {0.f,0.f,0.f,0.f,0.f,0.f,0.f,0.f,0.f,0.f};
  for (int q = tid; q < M; q += 128 * 256) {
    int hw = q >> 6, c = q & 63;
    const __bf16* p = cat + ((size_t)b * HW + hw) * 256 + c;
    float x0 = (float)p[0], x1 = (float)p[64], x2 = (float)p[128], x3 = (float)p[192];
    s[0] += x0; s[1] += x1; s[2] += x2; s[3] += x3;
    g[0] += x0*x0; g[1] += x0*x1; g[2] += x0*x2; g[3] += x0*x3;
    g[4] += x1*x1; g[5] += x1*x2; g[6] += x1*x3;
    g[7] += x2*x2; g[8] += x2*x3; g[9] += x3*x3;
  }
  #pragma unroll
  for (int q = 0; q < 4; ++q) {
    float v = s[q];
    for (int off = 16; off; off >>= 1) v += __shfl_xor(v, off, 32);
    if ((threadIdx.x & 31) == 0) atomicAdd(&stats[b * 16 + q], v);
  }
  #pragma unroll
  for (int q = 0; q < 10; ++q) {
    float v = g[q];
    for (int off = 16; off; off >>= 1) v += __shfl_xor(v, off, 32);
    if ((threadIdx.x & 31) == 0) atomicAdd(&stats[64 + b * 16 + q], v);
  }
}

// ---------------------------------------------------------------------------
// Covariance -> GLU attention -> fold into fuse1 weights, written directly in
// packed per-lane A order: q = ((((b*4)+coT)*8 + kk2)*32 + lane)*16 + e
// ---------------------------------------------------------------------------
__global__ void k_att_weff(const float* __restrict__ stats,
                           const float* __restrict__ alpha,
                           const float* __restrict__ gammav,
                           const float* __restrict__ betav,
                           const float* __restrict__ wf1,   // (64,256)
                           __bf16* __restrict__ weff,
                           int it)
{
  __shared__ float att_s[4][16];
  if (threadIdx.x < 4) {
    int b = threadIdx.x;
    const float Mf = (float)(CC * HW);
    float al = alpha[it], gm = gammav[it], be = betav[it];
    float mean[4], cov[4][4];
    for (int n = 0; n < 4; ++n) mean[n] = stats[b * 16 + n] / Mf;
    for (int n = 0; n < 4; ++n)
      for (int k = 0; k < 4; ++k) {
        int lo = n < k ? n : k, hia = n < k ? k : n;
        int idx = lo * (7 - lo) / 2 + hia;
        float gr = stats[64 + b * 16 + idx];
        cov[n][k] = al * (gr - Mf * mean[n] * mean[k]) / (Mf - 1.f);
      }
    float ms = 0.f;
    for (int n = 0; n < 4; ++n)
      for (int k = 0; k < 4; ++k) ms += cov[n][k] * cov[n][k];
    float inv = rsqrtf(ms / 16.f + 1e-5f);
    for (int n = 0; n < 4; ++n)
      for (int k = 0; k < 4; ++k) {
        float gv  = gm * cov[n][k] * inv + be;
        float sig = 1.f / (1.f + expf(-gv));
        att_s[b][n * 4 + k] = gv * sig;
      }
  }
  __syncthreads();
  for (int q = threadIdx.x; q < 4 * 64 * 256; q += blockDim.x) {
    int e = q & 15, lane = (q >> 4) & 31, kk2 = (q >> 9) & 7;
    int coT = (q >> 12) & 3, b = q >> 14;
    int m = lane & 15, hi = lane >> 4;
    int ci = kk2 * 32 + a_kmap(e, hi);
    int co = coT * 16 + m;
    int k = ci >> 6, cc = ci & 63;
    float acc = 0.f;
    #pragma unroll
    for (int n = 0; n < 4; ++n) {
      float a = att_s[b][n * 4 + k] + (n == k ? 1.f : 0.f);
      acc += wf1[co * 256 + n * 64 + cc] * a;
    }
    weff[q] = (__bf16)acc;
  }
}

// ---------------------------------------------------------------------------
// fuse1: 1x1 GEMM K=256 (attention pre-folded), lrelu -> bf16 y1 (haloed).
// ---------------------------------------------------------------------------
__global__ void k_fuse1(const __bf16* __restrict__ cat,
                        const __bf16* __restrict__ weff,
                        __bf16* __restrict__ y1h)   // (B,170,170,64) halo 5
{
  const int lane = threadIdx.x & 31;
  const int wave = blockIdx.x * WPB + (threadIdx.x >> 5);
  const int tilesPerB = 4 * 800;
  const int b    = wave / tilesPerB;
  const int r0   = wave % tilesPerB;
  const int coT  = r0 / 800;
  const int pixT = r0 % 800;
  const int nIdx = lane & 15, hi = lane >> 4;
  const int p0 = pixT * 32 + nIdx;

  const __bf16* bp0 = cat + ((size_t)b * HW + p0) * 256;
  const __bf16* bp1 = bp0 + 16 * 256;
  const __bf16* wl  = weff + (((size_t)b * 4 + coT) * 8 * 32 + lane) * 16;

  v8f acc0 = {}, acc1 = {};
  #pragma unroll
  for (int kk2 = 0; kk2 < 8; ++kk2) {
    if (kk2 == 0) __builtin_prefetch(bp0 + 32 * 256, 0, 1);
    v16bf Am = *(const v16bf*)(wl + kk2 * 512);
    int ci0 = kk2 * 32 + (hi << 4);
    v16bf B0 = *(const v16bf*)(bp0 + ci0);
    v16bf B1 = *(const v16bf*)(bp1 + ci0);
    acc0 = __builtin_amdgcn_wmma_f32_16x16x32_bf16(false, Am, false, B0,
                                                   (short)0, acc0, false, false);
    acc1 = __builtin_amdgcn_wmma_f32_16x16x32_bf16(false, Am, false, B1,
                                                   (short)0, acc1, false, false);
  }
  const int c0 = coT * 16 + 8 * hi;
  v8bf o0, o1;
  #pragma unroll
  for (int r = 0; r < 8; ++r) { o0[r] = (__bf16)lrelu(acc0[r]); o1[r] = (__bf16)lrelu(acc1[r]); }
  #pragma unroll
  for (int u = 0; u < 2; ++u) {
    int p = p0 + u * 16;
    int h = p / WW, w = p % WW;
    *(v8bf*)(y1h + (((size_t)b * YP + h + 5) * YP + (w + 5)) * CC + c0) = (u == 0) ? o0 : o1;
  }
}

// ---------------------------------------------------------------------------
// fuse2: 3x3 dilated (d=5) conv on haloed y1 (taps = immediate offsets, no
// selects) + residual; writes buf (NHWC f32, in place) and the mac2sai-
// permuted buf1 into the padded-view layout for next iteration's block_s.
// ---------------------------------------------------------------------------
__global__ void k_fuse2(const __bf16* __restrict__ y1h,   // haloed bf16
                        const __bf16* __restrict__ wpk,   // packed A
                        float* __restrict__ buf,          // NHWC f32
                        float* __restrict__ buf1p)        // padded views f32
{
  const int lane = threadIdx.x & 31;
  const int wave = blockIdx.x * WPB + (threadIdx.x >> 5);
  const int tilesPerB = 4 * 800;
  const int b    = wave / tilesPerB;
  const int r0   = wave % tilesPerB;
  const int coT  = r0 / 800;
  const int pixT = r0 % 800;
  const int nIdx = lane & 15, hi = lane >> 4;

  int h[2], w[2];
  const __bf16* base[2];
  #pragma unroll
  for (int u = 0; u < 2; ++u) {
    int p = pixT * 32 + nIdx + u * 16;
    h[u] = p / WW; w[u] = p % WW;
    base[u] = y1h + (((size_t)b * YP + h[u] + 5) * YP + (w[u] + 5)) * CC;
  }
  const __bf16* wl = wpk + ((size_t)coT * 9 * 2 * 32 + lane) * 16;

  v8f acc0 = {}, acc1 = {};
  #pragma unroll
  for (int tap = 0; tap < 9; ++tap) {
    const int off = ((tap / 3 - 1) * 5 * YP + (tap % 3 - 1) * 5) * CC;
    #pragma unroll
    for (int kk2 = 0; kk2 < 2; ++kk2) {
      v16bf Am = *(const v16bf*)(wl + (tap * 2 + kk2) * 512);
      int ci0 = kk2 * 32 + (hi << 4);
      v16bf B0 = *(const v16bf*)(base[0] + off + ci0);
      v16bf B1 = *(const v16bf*)(base[1] + off + ci0);
      acc0 = __builtin_amdgcn_wmma_f32_16x16x32_bf16(false, Am, false, B0,
                                                     (short)0, acc0, false, false);
      acc1 = __builtin_amdgcn_wmma_f32_16x16x32_bf16(false, Am, false, B1,
                                                     (short)0, acc1, false, false);
    }
  }
  const int c0 = coT * 16 + 8 * hi;
  #pragma unroll
  for (int u = 0; u < 2; ++u) {
    const v8f& a = (u == 0) ? acc0 : acc1;
    float4* bo = (float4*)(buf + (((size_t)b * HH + h[u]) * WW + w[u]) * CC + c0);
    float4 r0v = bo[0], r1v = bo[1];
    r0v.x += a[0]; r0v.y += a[1]; r0v.z += a[2]; r0v.w += a[3];
    r1v.x += a[4]; r1v.y += a[5]; r1v.z += a[6]; r1v.w += a[7];
    bo[0] = r0v; bo[1] = r1v;
    // mac2sai: (h,w) -> SAI row (h%32)*5 + h/32 -> view (bi,bj), in-view (i,j)
    int h1 = (h[u] & 31) * 5 + (h[u] >> 5);
    int w1 = (w[u] & 31) * 5 + (w[u] >> 5);
    int bi = h1 >> 5, ii = h1 & 31, bj = w1 >> 5, jj = w1 & 31;
    float4* b1 = (float4*)(buf1p + (((size_t)b * 25 + bi * 5 + bj) * VP
                                    + (ii + 1) * 34 + (jj + 1)) * CC + c0);
    b1[0] = r0v; b1[1] = r1v;
  }
}

// ---------------------------------------------------------------------------
extern "C" void kernel_launch(void* const* d_in, const int* in_sizes, int n_in,
                              void* d_out, int out_size, void* d_ws, size_t ws_size,
                              hipStream_t stream)
{
  (void)in_sizes; (void)n_in; (void)out_size; (void)ws_size;
  const float* x    = (const float*)d_in[0];
  const float* x1   = (const float*)d_in[1];
  const float* wspa = (const float*)d_in[2];
  const float* wang = (const float*)d_in[3];
  const float* wep1 = (const float*)d_in[4];
  const float* wep2 = (const float*)d_in[5];
  const float* alp  = (const float*)d_in[6];
  const float* gam  = (const float*)d_in[7];
  const float* bet  = (const float*)d_in[8];
  const float* wf1  = (const float*)d_in[9];
  const float* wf2  = (const float*)d_in[10];

  char* ws = (char*)d_ws;
  float*  buf    = (float*) (ws + 0);            // NHWC f32    26,214,400 B
  float*  buf1p  = (float*) (ws + 26214400);     // padded SAI  29,593,600 B
  __bf16* cat    = (__bf16*)(ws + 55808000);     // NHWC-256    52,428,800 B
  __bf16* tmp    = (__bf16*)(ws + 108236800);    // EPI tmp      2,621,440 B
  __bf16* y1h    = (__bf16*)(ws + 110858240);    // haloed      14,796,800 B
  float*  stats  = (float*) (ws + 125655040);    //                    512 B
  __bf16* weff   = (__bf16*)(ws + 125655552);    // packed         131,072 B
  __bf16* pkspa  = (__bf16*)(ws + 125786624);    //                 73,728 B
  __bf16* pkang  = (__bf16*)(ws + 125860352);    //                 73,728 B
  __bf16* pkep1  = (__bf16*)(ws + 125934080);    //                122,880 B
  __bf16* pkep2  = (__bf16*)(ws + 126056960);    //                 40,960 B
  __bf16* pkfu2  = (__bf16*)(ws + 126097920);    //                 73,728 B

  // zero halos once per launch (interiors are fully overwritten each iter)
  hipMemsetAsync(buf1p, 0, 29593600, stream);
  hipMemsetAsync(y1h,   0, 14796800, stream);

  k_tr_in    <<<1600, TPB, 0, stream>>>(x,  buf);
  k_tr_in_sai<<<1600, TPB, 0, stream>>>(x1, buf1p);

  for (int it = 0; it < 4; ++it) {
    const float* ws_  = wspa + (size_t)it * CC * CC * 9;
    const float* wa_  = wang + (size_t)it * CC * CC * 9;
    const float* we1_ = wep1 + (size_t)it * CC * CC * 15;
    const float* we2_ = wep2 + (size_t)it * 320 * CC;
    const float* wf1_ = wf1  + (size_t)it * CC * 256;
    const float* wf2_ = wf2  + (size_t)it * CC * CC * 9;

    k_pack<<<144, TPB, 0, stream>>>(ws_,  pkspa, 9,  36864);
    k_pack<<<144, TPB, 0, stream>>>(wa_,  pkang, 9,  36864);
    k_pack<<<240, TPB, 0, stream>>>(we1_, pkep1, 15, 61440);
    k_pack<<<80,  TPB, 0, stream>>>(we2_, pkep2, 1,  20480);
    k_pack<<<144, TPB, 0, stream>>>(wf2_, pkfu2, 9,  36864);

    k_conv_spa<<<1600, TPB, 0, stream>>>(buf1p, pkspa, cat);
    k_conv_ang<<<1600, TPB, 0, stream>>>(buf,   pkang, cat);
    k_epi1 <<<320,  TPB, 0, stream>>>(buf, pkep1, tmp, 0);
    k_epi2 <<<1600, TPB, 0, stream>>>(tmp, pkep2, cat, 0);
    k_epi1 <<<320,  TPB, 0, stream>>>(buf, pkep1, tmp, 1);
    k_epi2 <<<1600, TPB, 0, stream>>>(tmp, pkep2, cat, 1);
    k_zero <<<1, 128, 0, stream>>>(stats, 128);
    k_stats<<<512, TPB, 0, stream>>>(cat, stats);
    k_att_weff<<<1, TPB, 0, stream>>>(stats, alp, gam, bet, wf1_, weff, it);
    k_fuse1<<<1600, TPB, 0, stream>>>(cat, weff, y1h);
    k_fuse2<<<1600, TPB, 0, stream>>>(y1h, pkfu2, buf, buf1p);
  }

  k_tr_out<<<1600, TPB, 0, stream>>>(buf, (float*)d_out);
}
